// AttentionBlock_28673201668256
// MI455X (gfx1250) — matrix operand order repaired and checked by reading, not compiled, not run on hardware
//
#include <hip/hip_runtime.h>
#include <hip/hip_bf16.h>

#define B_  2
#define T_  2048
#define D_  2048
#define H_  32
#define HD_ 64
#define M_  (B_ * T_)   // 4096 rows of flattened (b,t)

// ---- CDNA5 WMMA types -------------------------------------------------------
typedef __attribute__((ext_vector_type(16))) __bf16       v16bf;
typedef __attribute__((ext_vector_type(8)))  float        v8f;
typedef __attribute__((ext_vector_type(4)))  unsigned int u32x4;
typedef __attribute__((ext_vector_type(4)))  int          v4i;

union FragU { v16bf v; u32x4 u[2]; };

// two 16-byte chunks (A-operand style: lane<16 -> K{0..7},{16..23})
__device__ __forceinline__ v16bf ldfrag2(const __bf16* p0, const __bf16* p1) {
  FragU f;
  f.u[0] = *(const u32x4*)p0;
  f.u[1] = *(const u32x4*)p1;
  return f.v;
}
// 32 contiguous bytes (contiguous k order: lane<16 -> K{0..15})
__device__ __forceinline__ v16bf ldfrag32(const __bf16* p) {
  FragU f;
  f.u[0] = *(const u32x4*)p;
  f.u[1] = *(const u32x4*)(p + 8);
  return f.v;
}

__device__ __forceinline__ v8f wmma_bf16(v16bf a, v16bf b, v8f c) {
  return __builtin_amdgcn_wmma_f32_16x16x32_bf16(
      /*neg_a=*/false, a, /*neg_b=*/false, b,
      /*c_mod=*/(short)0, c, /*reuse_a=*/false, /*reuse_b=*/false);
}

// ---- CDNA5 async global->LDS path (guarded; fallback = load + ds_store) -----
#if __has_builtin(__builtin_amdgcn_global_load_async_to_lds_b128) && \
    __has_builtin(__builtin_amdgcn_s_wait_asynccnt)
#define HAVE_ASYNC 1
#else
#define HAVE_ASYNC 0
#endif

typedef v4i __attribute__((address_space(1)))* as1_v4i;
typedef v4i __attribute__((address_space(3)))* as3_v4i;

__device__ __forceinline__ void async_copy_b128(const __bf16* g, __bf16* l) {
#if HAVE_ASYNC
  __builtin_amdgcn_global_load_async_to_lds_b128((as1_v4i)g, (as3_v4i)l, 0, 0);
#else
  u32x4 r = *(const u32x4*)g;
  *(u32x4*)l = r;
#endif
}
__device__ __forceinline__ void async_wait_all() {
#if HAVE_ASYNC
  __builtin_amdgcn_s_wait_asynccnt(0);
#endif
}

// ---- fp32 -> bf16 conversion ------------------------------------------------
__global__ __launch_bounds__(256) void cvt_f32_bf16(const float* __restrict__ src,
                                                    __bf16* __restrict__ dst, int n4) {
  int i = blockIdx.x * blockDim.x + threadIdx.x;
  if (i >= n4) return;
  float4 v = ((const float4*)src)[i];
  union { __bf16 h[4]; unsigned long long q; } u;
  u.h[0] = (__bf16)v.x; u.h[1] = (__bf16)v.y;
  u.h[2] = (__bf16)v.z; u.h[3] = (__bf16)v.w;
  ((unsigned long long*)dst)[i] = u.q;
}

// ---- bf16 WMMA GEMM with LDS double buffering -------------------------------
// C[m,n] = sum_k A[m,k] * W[n,k]
// mode 0: store bf16 as [B,H,T,HD]   (Q, K)
// mode 1: store bf16 as [B,H,HD,T]   (V transposed)
// mode 2: store fp32 as [M,N]        (final output)
#define LROW 40                      // padded row stride in bf16 elems (80 B)
#define TILE_E (128 * LROW)          // 5120 elems per A or B tile
#define BUF_E  (2 * TILE_E)          // elems per buffer (A then B)

__global__ __launch_bounds__(256)
void gemm_bf16_wmma(const __bf16* __restrict__ A, const __bf16* __restrict__ W,
                    void* __restrict__ Cout, int Ndim, int Kd, int mode) {
  __shared__ __bf16 smem[2 * BUF_E];           // 40 KB, double buffered

  const int tid  = threadIdx.x;
  const int lane = tid & 31;
  const int wave = tid >> 5;
  const int hi16 = lane >> 4;
  const int l16  = lane & 15;
  const int bm = blockIdx.x * 128 + (wave & 3) * 32;   // 4 waves along M
  const int bn = blockIdx.y * 128 + (wave >> 2) * 64;  // 2 waves along N

  // cooperative tile copy: each thread owns 32 B of A tile and 32 B of B tile
  const int lrow = tid >> 1;          // 0..127
  const int lcol = (tid & 1) * 16;    // element offset within 32-elem row
  const __bf16* gA = A + (size_t)(blockIdx.x * 128 + lrow) * Kd + lcol;
  const __bf16* gB = W + (size_t)(blockIdx.y * 128 + lrow) * Kd + lcol;
  const int ldst = lrow * LROW + lcol;

  v8f acc[2][4];
#pragma unroll
  for (int mf = 0; mf < 2; ++mf)
#pragma unroll
    for (int nf = 0; nf < 4; ++nf)
      acc[mf][nf] = (v8f){0.f, 0.f, 0.f, 0.f, 0.f, 0.f, 0.f, 0.f};

  // LDS fragment bases (per wave/lane)
  const int arow0 = (wave & 3) * 32 + l16;
  const int brow0 = (wave >> 2) * 64 + l16;

  // prologue: stage tile 0
  {
    __bf16* dA = smem + ldst;
    __bf16* dB = smem + TILE_E + ldst;
    async_copy_b128(gA,     dA);
    async_copy_b128(gA + 8, dA + 8);
    async_copy_b128(gB,     dB);
    async_copy_b128(gB + 8, dB + 8);
  }

  const int niter = Kd >> 5;
  for (int it = 0; it < niter; ++it) {
    const int buf = (it & 1) * BUF_E;
    async_wait_all();
    __syncthreads();

    // read fragments for this tile out of LDS (padded stride: conflict-free)
    v16bf a[2], b[4];
#pragma unroll
    for (int mf = 0; mf < 2; ++mf) {
      const __bf16* p = smem + buf + (arow0 + mf * 16) * LROW + hi16 * 8;
      a[mf] = ldfrag2(p, p + 16);
    }
#pragma unroll
    for (int nf = 0; nf < 4; ++nf) {
      const __bf16* p = smem + buf + TILE_E + (brow0 + nf * 16) * LROW + hi16 * 8;
      b[nf] = ldfrag2(p, p + 16);
    }

    // stage next tile into the other buffer (overlaps with WMMA below)
    if (it + 1 < niter) {
      const int nbuf = ((it + 1) & 1) * BUF_E;
      const int k0n  = (it + 1) << 5;
      __bf16* dA = smem + nbuf + ldst;
      __bf16* dB = smem + nbuf + TILE_E + ldst;
      async_copy_b128(gA + k0n,     dA);
      async_copy_b128(gA + k0n + 8, dA + 8);
      async_copy_b128(gB + k0n,     dB);
      async_copy_b128(gB + k0n + 8, dB + 8);
    }

#pragma unroll
    for (int mf = 0; mf < 2; ++mf)
#pragma unroll
      for (int nf = 0; nf < 4; ++nf)
        acc[mf][nf] = wmma_bf16(a[mf], b[nf], acc[mf][nf]);
  }

  // C layout: lane<16 -> M=r, lane>=16 -> M=8+r; N = lane&15
#pragma unroll
  for (int mf = 0; mf < 2; ++mf)
#pragma unroll
    for (int nf = 0; nf < 4; ++nf)
#pragma unroll
      for (int r = 0; r < 8; ++r) {
        int m = bm + mf * 16 + hi16 * 8 + r;
        int n = bn + nf * 16 + l16;
        float val = acc[mf][nf][r];
        if (mode == 2) {
          ((float*)Cout)[(size_t)m * Ndim + n] = val;
        } else {
          int bb = m / T_, t = m % T_, h = n >> 6, hd = n & 63;
          size_t idx = (mode == 0)
                           ? ((size_t)(bb * H_ + h) * T_ + t) * HD_ + hd
                           : ((size_t)(bb * H_ + h) * HD_ + hd) * T_ + t;
          ((__bf16*)Cout)[idx] = (__bf16)val;
        }
      }
}

// ---- RoPE on [B,H,T,HD] bf16 ------------------------------------------------
__global__ __launch_bounds__(256)
void rope_bf16(__bf16* __restrict__ qk, const float* __restrict__ cosT,
               const float* __restrict__ sinT, int total) {
  int i = blockIdx.x * blockDim.x + threadIdx.x;
  if (i >= total) return;
  int p  = i & 31;
  int t  = (i >> 5) & (T_ - 1);
  int bh = i >> 16;
  float c = cosT[t * 32 + p];
  float s = sinT[t * 32 + p];
  __bf16* ptr = qk + ((size_t)bh * T_ + t) * HD_ + p * 2;
  float xr = (float)ptr[0];
  float xi = (float)ptr[1];
  ptr[0] = (__bf16)(xr * c - xi * s);
  ptr[1] = (__bf16)(xr * s + xi * c);
}

// ---- Flash attention: one wave per (b,h, 16-query tile) ---------------------
__global__ __launch_bounds__(256)
void attn_wmma(const __bf16* __restrict__ Q, const __bf16* __restrict__ K,
               const __bf16* __restrict__ Vt, __bf16* __restrict__ Y) {
  const int lane = threadIdx.x & 31;
  const int wave = threadIdx.x >> 5;
  const int tile = blockIdx.x * 8 + wave;
  const int QT   = T_ / 16;
  const int qt   = tile % QT;
  const int bh   = tile / QT;
  const int hi16 = lane >> 4;
  const int l16  = lane & 15;

  const __bf16* Qb = Q  + (size_t)bh * T_ * HD_;
  const __bf16* Kb = K  + (size_t)bh * T_ * HD_;
  const __bf16* Vb = Vt + (size_t)bh * HD_ * T_;

  // Q as B operand of S^T = K * Q^T
  v16bf bq[2];
  {
    const __bf16* p = Qb + (size_t)(qt * 16 + l16) * HD_ + hi16 * 8;
    bq[0] = ldfrag2(p, p + 16);
    bq[1] = ldfrag2(p + 32, p + 48);
  }

  v8f o[4];
#pragma unroll
  for (int f = 0; f < 4; ++f) o[f] = (v8f){0.f,0.f,0.f,0.f,0.f,0.f,0.f,0.f};
  float m_run = -1e30f, l_run = 0.f;
  const int qg  = qt * 16 + l16;
  const int nkb = (qt + 2) >> 1;

  for (int kb = 0; kb < nkb; ++kb) {
    const int k0 = kb * 32;
    v8f s[2];
#pragma unroll
    for (int kf = 0; kf < 2; ++kf) {
      const __bf16* pk = Kb + (size_t)(k0 + kf * 16 + l16) * HD_ + hi16 * 8;
      v16bf ak0 = ldfrag2(pk, pk + 16);
      v16bf ak1 = ldfrag2(pk + 32, pk + 48);
      v8f t = (v8f){0.f,0.f,0.f,0.f,0.f,0.f,0.f,0.f};
      t = wmma_bf16(ak0, bq[0], t);
      t = wmma_bf16(ak1, bq[1], t);
      s[kf] = t;
    }
    float p0r[8], p1r[8];
    float mloc = -1e30f;
#pragma unroll
    for (int r = 0; r < 8; ++r) {
      int kg0 = k0 +      hi16 * 8 + r;
      int kg1 = k0 + 16 + hi16 * 8 + r;
      float v0 = s[0][r] * 0.125f; if (kg0 > qg) v0 = -1e30f;
      float v1 = s[1][r] * 0.125f; if (kg1 > qg) v1 = -1e30f;
      p0r[r] = v0; p1r[r] = v1;
      mloc = fmaxf(mloc, fmaxf(v0, v1));
    }
    mloc = fmaxf(mloc, __shfl_xor(mloc, 16, 32));
    float m_new = fmaxf(m_run, mloc);
    float alpha = __expf(m_run - m_new);
    float lsum = 0.f;
#pragma unroll
    for (int r = 0; r < 8; ++r) {
      p0r[r] = __expf(p0r[r] - m_new);
      p1r[r] = __expf(p1r[r] - m_new);
      lsum += p0r[r] + p1r[r];
    }
    lsum  += __shfl_xor(lsum, 16, 32);
    l_run  = l_run * alpha + lsum;
    m_run  = m_new;
#pragma unroll
    for (int f = 0; f < 4; ++f)
#pragma unroll
      for (int r = 0; r < 8; ++r) o[f][r] *= alpha;

    // rebuild P^T as B operand via lane-half exchange
    v16bf bp;
#pragma unroll
    for (int r = 0; r < 8; ++r) {
      float o0 = __shfl_xor(p0r[r], 16, 32);
      float o1 = __shfl_xor(p1r[r], 16, 32);
      float lo = hi16 ? o1 : p0r[r];
      float hi = hi16 ? p1r[r] : o0;
      bp[r]     = (__bf16)lo;
      bp[8 + r] = (__bf16)hi;
    }
#pragma unroll
    for (int f = 0; f < 4; ++f) {
      const __bf16* pv = Vb + (size_t)(f * 16 + l16) * T_ + k0 + hi16 * 16;
      v16bf av = ldfrag32(pv);
      o[f] = wmma_bf16(av, bp, o[f]);
    }
  }

  float inv = 1.0f / l_run;
  int bb = bh / H_, h = bh % H_;
  int t  = qt * 16 + l16;
#pragma unroll
  for (int f = 0; f < 4; ++f)
#pragma unroll
    for (int r = 0; r < 8; ++r) {
      int d = f * 16 + hi16 * 8 + r;
      Y[((size_t)(bb * T_ + t)) * D_ + h * HD_ + d] = (__bf16)(o[f][r] * inv);
    }
}

// ---- driver -----------------------------------------------------------------
extern "C" void kernel_launch(void* const* d_in, const int* in_sizes, int n_in,
                              void* d_out, int out_size, void* d_ws, size_t ws_size,
                              hipStream_t stream) {
  const float* x  = (const float*)d_in[0];
  const float* fc = (const float*)d_in[1];
  const float* fs = (const float*)d_in[2];
  const float* wq = (const float*)d_in[3];
  const float* wk = (const float*)d_in[4];
  const float* wv = (const float*)d_in[5];
  const float* wo = (const float*)d_in[6];
  float* out = (float*)d_out;

  char* ws = (char*)d_ws;
  size_t off = 0;
  const size_t xbytes = (size_t)M_ * D_ * sizeof(__bf16);
  const size_t wbytes = (size_t)D_ * D_ * sizeof(__bf16);
  __bf16* xb  = (__bf16*)(ws + off); off += xbytes;
  __bf16* wqb = (__bf16*)(ws + off); off += wbytes;
  __bf16* wkb = (__bf16*)(ws + off); off += wbytes;
  __bf16* wvb = (__bf16*)(ws + off); off += wbytes;
  __bf16* wob = (__bf16*)(ws + off); off += wbytes;
  __bf16* Qb  = (__bf16*)(ws + off); off += xbytes;   // [B,H,T,HD]
  __bf16* Kb  = (__bf16*)(ws + off); off += xbytes;   // [B,H,T,HD]
  __bf16* Vtb = (__bf16*)(ws + off); off += xbytes;   // [B,H,HD,T]
  __bf16* yb  = (__bf16*)(ws + off); off += xbytes;   // [M,D]

  {
    int n4x = M_ * D_ / 4, n4w = D_ * D_ / 4;
    cvt_f32_bf16<<<(n4x + 255) / 256, 256, 0, stream>>>(x,  xb,  n4x);
    cvt_f32_bf16<<<(n4w + 255) / 256, 256, 0, stream>>>(wq, wqb, n4w);
    cvt_f32_bf16<<<(n4w + 255) / 256, 256, 0, stream>>>(wk, wkb, n4w);
    cvt_f32_bf16<<<(n4w + 255) / 256, 256, 0, stream>>>(wv, wvb, n4w);
    cvt_f32_bf16<<<(n4w + 255) / 256, 256, 0, stream>>>(wo, wob, n4w);
  }

  dim3 gg(M_ / 128, D_ / 128);
  gemm_bf16_wmma<<<gg, 256, 0, stream>>>(xb, wqb, Qb,  D_, D_, 0);
  gemm_bf16_wmma<<<gg, 256, 0, stream>>>(xb, wkb, Kb,  D_, D_, 0);
  gemm_bf16_wmma<<<gg, 256, 0, stream>>>(xb, wvb, Vtb, D_, D_, 1);

  {
    int total = B_ * H_ * T_ * (HD_ / 2);
    rope_bf16<<<(total + 255) / 256, 256, 0, stream>>>(Qb, fc, fs, total);
    rope_bf16<<<(total + 255) / 256, 256, 0, stream>>>(Kb, fc, fs, total);
  }

  {
    int tiles = B_ * H_ * (T_ / 16);
    attn_wmma<<<tiles / 8, 256, 0, stream>>>(Qb, Kb, Vtb, yb);
  }

  gemm_bf16_wmma<<<gg, 256, 0, stream>>>(yb, wob, out, D_, D_, 2);
}
